// EGNNModel_28939489640554
// MI455X (gfx1250) — matrix operand order repaired and checked by reading, not compiled, not run on hardware
//
#include <hip/hip_runtime.h>

typedef __attribute__((ext_vector_type(16))) __bf16 v16bf;
typedef __attribute__((ext_vector_type(8)))  float  v8f;
typedef __attribute__((ext_vector_type(8)))  unsigned int v8u;

#define LAYERS 5

// ---------- helpers ----------
static __device__ __forceinline__ unsigned int f2bf(float f) {
    unsigned int u = __builtin_bit_cast(unsigned int, f);
    unsigned int r = (u + 0x7FFFu + ((u >> 16) & 1u)) >> 16;   // RNE
    return r & 0xFFFFu;
}
static __device__ __forceinline__ unsigned int pk_bf16(float x, float y) {
    return f2bf(x) | (f2bf(y) << 16);
}
static __device__ __forceinline__ v8f wmma_bf16(v8u a, v8u b, v8f c) {
    // D = A(16x32 bf16) * B(32x16 bf16) + C(16x16 f32)
    return __builtin_amdgcn_wmma_f32_16x16x32_bf16(
        false, __builtin_bit_cast(v16bf, a),
        false, __builtin_bit_cast(v16bf, b),
        (short)0, c, false, false);
}
static __device__ __forceinline__ v8f v8f_zero() {
    v8f z = {0.f,0.f,0.f,0.f,0.f,0.f,0.f,0.f};
    return z;
}

// ---------- prep kernels ----------
__global__ void init_h_kernel(const int* __restrict__ x, const float* __restrict__ emb,
                              float* __restrict__ h, unsigned short* __restrict__ hbf, int Nn) {
    int i = blockIdx.x * blockDim.x + threadIdx.x;
    if (i >= Nn * 128) return;
    int n = i >> 7, d = i & 127;
    float v = emb[(size_t)x[n] * 128 + d];
    h[i] = v;
    hbf[i] = (unsigned short)f2bf(v);
}

__global__ void deg_kernel(const int* __restrict__ dst, float* __restrict__ deg, int Ee) {
    int i = blockIdx.x * blockDim.x + threadIdx.x;
    if (i < Ee) atomicAdd(&deg[dst[i]], 1.0f);
}

// rbf[e][k] = exp(-gamma*(d-c_k)^2), c_k = k*30/63 ; stored as bf16 pairs (uint)
__global__ void rbf_kernel(const float* __restrict__ pos, const int* __restrict__ src,
                           const int* __restrict__ dst, unsigned int* __restrict__ rbf, int Ee) {
    int i = blockIdx.x * blockDim.x + threadIdx.x;
    if (i >= Ee * 32) return;
    int e = i >> 5, j = (i & 31) * 2;
    int s = src[e], t = dst[e];
    float dx = pos[(size_t)t*3+0] - pos[(size_t)s*3+0];
    float dy = pos[(size_t)t*3+1] - pos[(size_t)s*3+1];
    float dz = pos[(size_t)t*3+2] - pos[(size_t)s*3+2];
    float d  = sqrtf(dx*dx + dy*dy + dz*dz);
    const float width = 30.0f / 63.0f;
    const float gamma = 1.0f / (2.0f * width * width);
    float c0 = (float)j * width, c1 = (float)(j + 1) * width;
    float r0 = __expf(-gamma * (d - c0) * (d - c0));
    float r1 = __expf(-gamma * (d - c1) * (d - c1));
    rbf[(size_t)e * 32 + (j >> 1)] = pk_bf16(r0, r1);
}

// Fused edge weight, stored transposed bf16: WcatT[l][n][k], k<256 -> Wm rows, k>=256 -> (Wcp@Wc)
__global__ void fuse_wcat_kernel(const float* __restrict__ Wm, const float* __restrict__ Wcp,
                                 const float* __restrict__ Wc, unsigned short* __restrict__ WcatT) {
    int i = blockIdx.x * blockDim.x + threadIdx.x;
    if (i >= LAYERS * 128 * 320) return;
    int l = i / (128 * 320);
    int r = i % (128 * 320);
    int n = r / 320;   // output column
    int k = r % 320;   // input row
    float w;
    if (k < 256) {
        w = Wm[(size_t)l * 256 * 128 + (size_t)k * 128 + n];
    } else {
        const float* wcp = Wcp + (size_t)l * 64 * 128 + (size_t)(k - 256) * 128;
        const float* wc  = Wc  + (size_t)l * 128 * 128;
        float s = 0.f;
        for (int j = 0; j < 128; ++j) s = fmaf(wcp[j], wc[(size_t)j * 128 + n], s);
        w = s;
    }
    WcatT[(size_t)l * 128 * 320 + (size_t)n * 320 + k] = (unsigned short)f2bf(w);
}

__global__ void fuse_bias_kernel(const float* __restrict__ bm, const float* __restrict__ bcp,
                                 const float* __restrict__ Wc, float* __restrict__ biasf) {
    int i = blockIdx.x * blockDim.x + threadIdx.x;
    if (i >= LAYERS * 128) return;
    int l = i / 128, n = i % 128;
    float s = bm[(size_t)l * 128 + n];
    const float* b  = bcp + (size_t)l * 128;
    const float* wc = Wc  + (size_t)l * 128 * 128;
    for (int j = 0; j < 128; ++j) s = fmaf(b[j], wc[(size_t)j * 128 + n], s);
    biasf[i] = s;
}

__global__ void conv_wu1_kernel(const float* __restrict__ Wu1, unsigned short* __restrict__ Wu1T) {
    int i = blockIdx.x * blockDim.x + threadIdx.x;
    if (i >= LAYERS * 128 * 256) return;
    int l = i / (128 * 256);
    int r = i % (128 * 256);
    int n = r / 256, k = r % 256;
    Wu1T[(size_t)l * 128 * 256 + (size_t)n * 256 + k] =
        (unsigned short)f2bf(Wu1[(size_t)l * 256 * 128 + (size_t)k * 128 + n]);
}

__global__ void conv_wu2_kernel(const float* __restrict__ Wu2, unsigned short* __restrict__ Wu2T) {
    int i = blockIdx.x * blockDim.x + threadIdx.x;
    if (i >= LAYERS * 128 * 128) return;
    int l = i / (128 * 128);
    int r = i % (128 * 128);
    int n = r / 128, k = r % 128;
    Wu2T[(size_t)l * 128 * 128 + (size_t)n * 128 + k] =
        (unsigned short)f2bf(Wu2[(size_t)l * 128 * 128 + (size_t)k * 128 + n]);
}

// ---------- edge message kernel ----------
// msg = [h_dst|h_src|rbf](K=320) @ Wcat + biasf ; agg[dst] += msg
// 8 waves/block process the SAME 16-edge tile; wave w owns output columns [16w,16w+16).
// Its 10 B fragments (80 VGPRs) are loop-invariant and live in registers — no LDS, no spills.
__global__ void edge_msg_kernel(const unsigned short* __restrict__ hbf,      // [N][128] bf16
                                const unsigned short* __restrict__ rbfbf,    // [E][64] bf16
                                const int* __restrict__ srcIdx, const int* __restrict__ dstIdx,
                                const unsigned short* __restrict__ WcatT,    // [128][320] bf16 (W^T)
                                const float* __restrict__ biasf,             // [128]
                                float* __restrict__ agg, int Ee) {
    const int lane = threadIdx.x & 31;
    const int wave = threadIdx.x >> 5;        // owned N-tile
    const int m  = lane & 15;
    const int hh = lane >> 4;
    const int n  = wave * 16 + m;             // owned output column

    // B fragments for owned N-tile, K = 320 = 10 chunks of 32 (kept in VGPRs)
    const unsigned int* wrow = (const unsigned int*)WcatT + (size_t)n * 160;
    v8u B[10];
    #pragma unroll
    for (int kc = 0; kc < 10; ++kc) {
        #pragma unroll
        for (int v = 0; v < 8; ++v) B[kc][v] = wrow[kc * 16 + 8 * (v >> 2) + 4 * hh + (v & 3)];
    }
    const float bias = biasf[n];

    const int numTiles = (Ee + 15) >> 4;
    for (int tile = blockIdx.x; tile < numTiles; tile += gridDim.x) {
        if (tile + gridDim.x < numTiles) {    // global_prefetch_b8 for next tile's indices
            __builtin_prefetch(dstIdx + (size_t)(tile + gridDim.x) * 16, 0, 0);
            __builtin_prefetch(srcIdx + (size_t)(tile + gridDim.x) * 16, 0, 0);
        }
        int e  = tile * 16 + m;
        int ec = (e < Ee) ? e : (Ee - 1);
        int dnode = dstIdx[ec];
        int snode = srcIdx[ec];
        const unsigned int* hd = (const unsigned int*)(hbf + (size_t)dnode * 128);
        const unsigned int* hs = (const unsigned int*)(hbf + (size_t)snode * 128);
        const unsigned int* rb = (const unsigned int*)(rbfbf + (size_t)ec * 64);

        v8f acc = v8f_zero();
        #pragma unroll
        for (int kc = 0; kc < 10; ++kc) {
            const unsigned int* abase = (kc < 4) ? (hd + kc * 16)
                                      : (kc < 8) ? (hs + (kc - 4) * 16)
                                                 : (rb + (kc - 8) * 16);
            v8u a;
            #pragma unroll
            for (int v = 0; v < 8; ++v)       // two contiguous b128 loads per chunk
                a[v] = abase[8 * (v >> 2) + 4 * hh + (v & 3)];
            acc = wmma_bf16(a, B[kc], acc);
        }

        // scatter this wave's 16 columns; lane holds rows M = v + 8*hh at column n
        #pragma unroll
        for (int v = 0; v < 8; ++v) {
            int rowM = v + 8 * hh;
            int er = tile * 16 + rowM;
            if (er < Ee) {
                int dRow = __shfl(dnode, rowM, 32);
                atomicAdd(&agg[(size_t)dRow * 128 + n], acc[v] + bias);
            }
        }
    }
}

// ---------- fused node update ----------
// u = relu(LN([h|agg]@Wu1+bu1)) ; u = relu(LN(u@Wu2+bu2)) ; h += (deg>0 ? u : h)
// 8 waves/block cooperate on one 16-node tile; wave w owns columns [16w,16w+16).
// B1 (64 VGPRs) + B2 (32 VGPRs) in registers; LN partials + bf16 staging in 5KB static LDS.
__global__ void node_update_kernel(float* __restrict__ h, unsigned short* __restrict__ hbf,
                                   const float* __restrict__ agg,
                                   const unsigned short* __restrict__ Wu1T,  // [128][256] bf16
                                   const unsigned short* __restrict__ Wu2T,  // [128][128] bf16
                                   const float* __restrict__ bu1, const float* __restrict__ g1,
                                   const float* __restrict__ be1,
                                   const float* __restrict__ bu2, const float* __restrict__ g2,
                                   const float* __restrict__ be2,
                                   const float* __restrict__ deg, int Nn) {
    __shared__ unsigned short ustag[16 * 128];  // normalized u, shared A operand for GEMM2
    __shared__ float part_p[8][16];
    __shared__ float part_q[8][16];

    const int lane = threadIdx.x & 31;
    const int wave = threadIdx.x >> 5;
    const int m  = lane & 15;
    const int hh = lane >> 4;
    const int n  = wave * 16 + m;             // owned output column

    const unsigned int* w1row = (const unsigned int*)Wu1T + (size_t)n * 128;
    v8u B1[8];
    #pragma unroll
    for (int kc = 0; kc < 8; ++kc) {
        #pragma unroll
        for (int v = 0; v < 8; ++v) B1[kc][v] = w1row[kc * 16 + 8 * (v >> 2) + 4 * hh + (v & 3)];
    }
    const unsigned int* w2row = (const unsigned int*)Wu2T + (size_t)n * 64;
    v8u B2[4];
    #pragma unroll
    for (int kc = 0; kc < 4; ++kc) {
        #pragma unroll
        for (int v = 0; v < 8; ++v) B2[kc][v] = w2row[kc * 16 + 8 * (v >> 2) + 4 * hh + (v & 3)];
    }
    const float bu1v = bu1[n], g1v = g1[n], be1v = be1[n];
    const float bu2v = bu2[n], g2v = g2[n], be2v = be2[n];

    const unsigned int* stagu = (const unsigned int*)ustag;
    const int numTiles = (Nn + 15) >> 4;

    // block-uniform tile loop (all 8 waves share `tile`) -> barriers are safe
    for (int tile = blockIdx.x; tile < numTiles; tile += gridDim.x) {
        int node = tile * 16 + m;
        int nc = (node < Nn) ? node : (Nn - 1);
        const float* hrow = h   + (size_t)nc * 128;
        const float* arow = agg + (size_t)nc * 128;
        float degv = deg[nc];

        // -------- GEMM1: [h | agg](K=256) @ Wu1 (owned 16 columns) --------
        v8f acc = v8f_zero();
        #pragma unroll
        for (int kc = 0; kc < 8; ++kc) {
            const float* base = (kc < 4) ? (hrow + kc * 32) : (arow + (kc - 4) * 32);
            v8u a;
            #pragma unroll
            for (int v = 0; v < 8; ++v) {
                int K = 16 * (v >> 2) + 8 * hh + 2 * (v & 3);
                a[v] = pk_bf16(base[K], base[K + 1]);   // on-the-fly f32 -> bf16
            }
            acc = wmma_bf16(a, B1[kc], acc);
        }
        #pragma unroll
        for (int v = 0; v < 8; ++v) acc[v] += bu1v;

        // -------- LayerNorm1 (cross-wave): per-row partials over owned 16 cols --------
        #pragma unroll
        for (int v = 0; v < 8; ++v) {
            float p = acc[v], q = acc[v] * acc[v];
            #pragma unroll
            for (int off = 1; off < 16; off <<= 1) {    // stays within 16-lane half
                p += __shfl_xor(p, off, 32);
                q += __shfl_xor(q, off, 32);
            }
            if (m == 0) { part_p[wave][v + 8 * hh] = p; part_q[wave][v + 8 * hh] = q; }
        }
        __syncthreads();

        float mean[8], rstd[8];
        #pragma unroll
        for (int v = 0; v < 8; ++v) {
            int row = v + 8 * hh;
            float p = 0.f, q = 0.f;
            #pragma unroll
            for (int w = 0; w < 8; ++w) { p += part_p[w][row]; q += part_q[w][row]; }
            mean[v] = p * (1.f / 128.f);
            float var = q * (1.f / 128.f) - mean[v] * mean[v];
            rstd[v] = rsqrtf(var + 1e-5f);
        }
        // normalize + relu -> shared bf16 staging (row-major [16][128])
        #pragma unroll
        for (int v = 0; v < 8; ++v) {
            float xv = fmaxf((acc[v] - mean[v]) * rstd[v] * g1v + be1v, 0.f);
            ustag[(v + 8 * hh) * 128 + n] = (unsigned short)f2bf(xv);
        }
        __syncthreads();

        // -------- GEMM2: u(K=128) @ Wu2 (A from shared staging) --------
        v8f acc2 = v8f_zero();
        #pragma unroll
        for (int kc = 0; kc < 4; ++kc) {
            v8u a;
            #pragma unroll
            for (int v = 0; v < 8; ++v)
                a[v] = stagu[m * 64 + kc * 16 + 8 * (v >> 2) + 4 * hh + (v & 3)];
            acc2 = wmma_bf16(a, B2[kc], acc2);
        }
        #pragma unroll
        for (int v = 0; v < 8; ++v) acc2[v] += bu2v;

        // -------- LayerNorm2 (cross-wave) --------
        #pragma unroll
        for (int v = 0; v < 8; ++v) {
            float p = acc2[v], q = acc2[v] * acc2[v];
            #pragma unroll
            for (int off = 1; off < 16; off <<= 1) {
                p += __shfl_xor(p, off, 32);
                q += __shfl_xor(q, off, 32);
            }
            if (m == 0) { part_p[wave][v + 8 * hh] = p; part_q[wave][v + 8 * hh] = q; }
        }
        __syncthreads();
        #pragma unroll
        for (int v = 0; v < 8; ++v) {
            int row = v + 8 * hh;
            float p = 0.f, q = 0.f;
            #pragma unroll
            for (int w = 0; w < 8; ++w) { p += part_p[w][row]; q += part_q[w][row]; }
            mean[v] = p * (1.f / 128.f);
            float var = q * (1.f / 128.f) - mean[v] * mean[v];
            rstd[v] = rsqrtf(var + 1e-5f);
        }
        #pragma unroll
        for (int v = 0; v < 8; ++v)
            acc2[v] = fmaxf((acc2[v] - mean[v]) * rstd[v] * g2v + be2v, 0.f);

        // -------- masked residual: h = h + (deg>0 ? u : h); refresh bf16 mirror --------
        #pragma unroll
        for (int v = 0; v < 8; ++v) {
            int rowM = v + 8 * hh;
            int nd = tile * 16 + rowM;
            if (nd < Nn) {
                float dv = __shfl(degv, rowM, 32);
                float hold = h[(size_t)nd * 128 + n];
                float hupd = (dv > 0.f) ? acc2[v] : hold;
                float hnew = hold + hupd;
                h[(size_t)nd * 128 + n]   = hnew;
                hbf[(size_t)nd * 128 + n] = (unsigned short)f2bf(hnew);
            }
        }
        __syncthreads();   // protect ustag / partials before next tile reuses them
    }
}

// ---------- pooling + head ----------
__global__ void pool_kernel(const float* __restrict__ h, const int* __restrict__ batch,
                            float* __restrict__ pooled, int Nn) {
    int i = blockIdx.x * blockDim.x + threadIdx.x;
    if (i >= Nn * 128) return;
    int n = i >> 7, d = i & 127;
    atomicAdd(&pooled[(size_t)batch[n] * 128 + d], h[i]);
}

__global__ void head_kernel(const float* __restrict__ pooled, const float* __restrict__ Wp1,
                            const float* __restrict__ bp1, const float* __restrict__ gp,
                            const float* __restrict__ bpv, const float* __restrict__ Wp2,
                            const float* __restrict__ bp2, float* __restrict__ out) {
    __shared__ float row[128];
    __shared__ float red[128];
    int g = blockIdx.x, d = threadIdx.x;
    row[d] = pooled[(size_t)g * 128 + d];
    __syncthreads();
    float t = bp1[d];
    for (int k = 0; k < 128; ++k) t = fmaf(row[k], Wp1[(size_t)k * 128 + d], t);
    red[d] = t; __syncthreads();
    for (int s = 64; s > 0; s >>= 1) { if (d < s) red[d] += red[d + s]; __syncthreads(); }
    float mean = red[0] * (1.f / 128.f);
    __syncthreads();
    float c = t - mean;
    red[d] = c * c; __syncthreads();
    for (int s = 64; s > 0; s >>= 1) { if (d < s) red[d] += red[d + s]; __syncthreads(); }
    float var = red[0] * (1.f / 128.f);
    __syncthreads();
    float z = fmaxf(c * rsqrtf(var + 1e-5f) * gp[d] + bpv[d], 0.f);
    red[d] = z * Wp2[d]; __syncthreads();
    for (int s = 64; s > 0; s >>= 1) { if (d < s) red[d] += red[d + s]; __syncthreads(); }
    if (d == 0) out[g] = red[0] + bp2[0];
}

// ---------- host launcher ----------
extern "C" void kernel_launch(void* const* d_in, const int* in_sizes, int n_in,
                              void* d_out, int out_size, void* d_ws, size_t ws_size,
                              hipStream_t stream) {
    const int*   x     = (const int*)  d_in[0];
    const float* pos   = (const float*)d_in[1];
    const int*   eidx  = (const int*)  d_in[2];
    const int*   batch = (const int*)  d_in[3];
    const float* emb   = (const float*)d_in[4];
    const float* Wcp   = (const float*)d_in[5];
    const float* bcp   = (const float*)d_in[6];
    const float* Wm    = (const float*)d_in[7];
    const float* bm    = (const float*)d_in[8];
    const float* Wc    = (const float*)d_in[9];
    const float* Wu1   = (const float*)d_in[10];
    const float* bu1   = (const float*)d_in[11];
    const float* g1    = (const float*)d_in[12];
    const float* be1   = (const float*)d_in[13];
    const float* Wu2   = (const float*)d_in[14];
    const float* bu2   = (const float*)d_in[15];
    const float* g2    = (const float*)d_in[16];
    const float* be2   = (const float*)d_in[17];
    const float* Wp1   = (const float*)d_in[18];
    const float* bp1   = (const float*)d_in[19];
    const float* gp    = (const float*)d_in[20];
    const float* bpv   = (const float*)d_in[21];
    const float* Wp2   = (const float*)d_in[22];
    const float* bp2   = (const float*)d_in[23];

    const int Nn = in_sizes[0];
    const int Ee = in_sizes[2] / 2;
    const int* src = eidx;
    const int* dst = eidx + Ee;

    char* ws = (char*)d_ws;
    size_t off = 0;
    auto alloc = [&](size_t bytes) -> void* {
        void* p = ws + off;
        off = (off + bytes + 255) & ~(size_t)255;
        return p;
    };
    float*          h      = (float*)         alloc((size_t)Nn * 128 * 4);
    unsigned short* hbf    = (unsigned short*)alloc((size_t)Nn * 128 * 2);
    float*          agg    = (float*)         alloc((size_t)Nn * 128 * 4);
    unsigned int*   rbfbf  = (unsigned int*)  alloc((size_t)Ee * 32 * 4);
    float*          deg    = (float*)         alloc((size_t)Nn * 4);
    unsigned short* WcatT  = (unsigned short*)alloc((size_t)LAYERS * 128 * 320 * 2);
    float*          biasf  = (float*)         alloc((size_t)LAYERS * 128 * 4);
    unsigned short* Wu1T   = (unsigned short*)alloc((size_t)LAYERS * 128 * 256 * 2);
    unsigned short* Wu2T   = (unsigned short*)alloc((size_t)LAYERS * 128 * 128 * 2);
    float*          pooled = (float*)         alloc((size_t)64 * 128 * 4);

    hipMemsetAsync(deg, 0, (size_t)Nn * 4, stream);
    hipMemsetAsync(pooled, 0, (size_t)64 * 128 * 4, stream);

    init_h_kernel<<<(Nn * 128 + 255) / 256, 256, 0, stream>>>(x, emb, h, hbf, Nn);
    deg_kernel<<<(Ee + 255) / 256, 256, 0, stream>>>(dst, deg, Ee);
    rbf_kernel<<<(Ee * 32 + 255) / 256, 256, 0, stream>>>(pos, src, dst, rbfbf, Ee);
    fuse_wcat_kernel<<<(LAYERS * 128 * 320 + 255) / 256, 256, 0, stream>>>(Wm, Wcp, Wc, WcatT);
    fuse_bias_kernel<<<(LAYERS * 128 + 127) / 128, 128, 0, stream>>>(bm, bcp, Wc, biasf);
    conv_wu1_kernel<<<(LAYERS * 128 * 256 + 255) / 256, 256, 0, stream>>>(Wu1, Wu1T);
    conv_wu2_kernel<<<(LAYERS * 128 * 128 + 255) / 256, 256, 0, stream>>>(Wu2, Wu2T);

    const int eTiles = (Ee + 15) / 16;
    const int nTiles = (Nn + 15) / 16;
    const int eBlocks = (eTiles < 2048) ? eTiles : 2048;   // persistent blocks
    const int nBlocks = (nTiles < 1024) ? nTiles : 1024;

    for (int l = 0; l < LAYERS; ++l) {
        hipMemsetAsync(agg, 0, (size_t)Nn * 128 * 4, stream);
        edge_msg_kernel<<<eBlocks, 256, 0, stream>>>(
            hbf, (const unsigned short*)rbfbf, src, dst,
            WcatT + (size_t)l * 128 * 320, biasf + (size_t)l * 128, agg, Ee);
        node_update_kernel<<<nBlocks, 256, 0, stream>>>(
            h, hbf, agg,
            Wu1T + (size_t)l * 128 * 256, Wu2T + (size_t)l * 128 * 128,
            bu1 + (size_t)l * 128, g1 + (size_t)l * 128, be1 + (size_t)l * 128,
            bu2 + (size_t)l * 128, g2 + (size_t)l * 128, be2 + (size_t)l * 128,
            deg, Nn);
    }

    pool_kernel<<<(Nn * 128 + 255) / 256, 256, 0, stream>>>(h, batch, pooled, Nn);
    head_kernel<<<64, 128, 0, stream>>>(pooled, Wp1, bp1, gp, bpv, Wp2, bp2, (float*)d_out);
}
